// FusedDeepSeekMoEMLP_76141180223863
// MI455X (gfx1250) — matrix operand order repaired and checked by reading, not compiled, not run on hardware
//
#include <hip/hip_runtime.h>
#include <hip/hip_bf16.h>

// FusedDeepSeekMoEMLP for MI455X (gfx1250): sparse top-2 MoE with bf16 WMMA.
// d_out = [out (4096*1024 f32) | lb_loss | z_loss | imp_loss | tokens_per_expert(8)]

#define DIMD   1024
#define HDIMD  4096
#define NEXP   8
#define NTOKS  4096          // B*T
#define TM     32            // tokens per workgroup tile
#define BNH    64            // hidden-dim block width

typedef __attribute__((ext_vector_type(16))) __bf16 bf16x16;
typedef __attribute__((ext_vector_type(8)))  __bf16 bf16x8;
typedef __attribute__((ext_vector_type(8)))  float  f32x8;
typedef __attribute__((ext_vector_type(4)))  float  f32x4;

union ABFrag { bf16x16 v; bf16x8 h[2]; __bf16 e[16]; };

// ---------------------------------------------------------------------------
// Kernel 0: zero the accumulated output region + scratch accumulators
// ---------------------------------------------------------------------------
__global__ __launch_bounds__(256) void moe_zero_kernel(f32x4* __restrict__ out4,
                                                       int* __restrict__ accum) {
    size_t i = ((size_t)blockIdx.x * 256 + threadIdx.x) * 2;
    out4[i]     = (f32x4)0.0f;
    out4[i + 1] = (f32x4)0.0f;
    if (blockIdx.x == 0 && threadIdx.x < 64) accum[threadIdx.x] = 0;
}

// ---------------------------------------------------------------------------
// Kernel 1: router — logits, softmax, top-2, aux-loss accumulators,
// per-expert token lists, and x -> bf16 conversion (fused).
// One wave32 per token, 8 waves / block.
// ---------------------------------------------------------------------------
__global__ __launch_bounds__(256) void moe_router_kernel(
    const float* __restrict__ x, const float* __restrict__ rw,
    __bf16* __restrict__ xb, int* __restrict__ cnt,
    int* __restrict__ list, float* __restrict__ wgt,
    float* __restrict__ imp, float* __restrict__ tpe,
    float* __restrict__ zacc) {
    __shared__ float srw[NEXP][DIMD];
    for (int i = threadIdx.x; i < NEXP * DIMD; i += 256)
        srw[i >> 10][i & (DIMD - 1)] = rw[i];
    __syncthreads();

    const int lane = threadIdx.x & 31;
    const int wv   = threadIdx.x >> 5;
    const int tok  = blockIdx.x * 8 + wv;

    float acc[NEXP];
#pragma unroll
    for (int e = 0; e < NEXP; ++e) acc[e] = 0.0f;

    const float* xr = x + (size_t)tok * DIMD;
    for (int j = lane; j < DIMD; j += 32) {
        float xv = xr[j];
        xb[(size_t)tok * DIMD + j] = (__bf16)xv;   // fused fp32->bf16 convert
#pragma unroll
        for (int e = 0; e < NEXP; ++e) acc[e] += xv * srw[e][j];
    }
#pragma unroll
    for (int e = 0; e < NEXP; ++e) {
        float s = acc[e];
#pragma unroll
        for (int o = 16; o > 0; o >>= 1) s += __shfl_xor(s, o, 32);
        acc[e] = s;
    }

    if (lane == 0) {
        float m = acc[0];
#pragma unroll
        for (int e = 1; e < NEXP; ++e) m = fmaxf(m, acc[e]);
        float p[NEXP], se = 0.0f;
#pragma unroll
        for (int e = 0; e < NEXP; ++e) { p[e] = __expf(acc[e] - m); se += p[e]; }
        float inv = 1.0f / se;
#pragma unroll
        for (int e = 0; e < NEXP; ++e) p[e] *= inv;
        float lse = m + __logf(se);
        atomicAdd(zacc, lse * lse);
#pragma unroll
        for (int e = 0; e < NEXP; ++e) atomicAdd(&imp[e], p[e]);

        int i0 = 0;
#pragma unroll
        for (int e = 1; e < NEXP; ++e) if (p[e] > p[i0]) i0 = e;
        int i1 = (i0 == 0) ? 1 : 0;
#pragma unroll
        for (int e = 0; e < NEXP; ++e) if (e != i0 && p[e] > p[i1]) i1 = e;

        float w0 = p[i0], w1 = p[i1], s2 = w0 + w1;
        w0 /= s2; w1 /= s2;
        atomicAdd(&tpe[i0], 1.0f);
        int p0 = atomicAdd(&cnt[i0], 1);
        list[i0 * NTOKS + p0] = tok; wgt[i0 * NTOKS + p0] = w0;
        int p1 = atomicAdd(&cnt[i1], 1);
        list[i1 * NTOKS + p1] = tok; wgt[i1 * NTOKS + p1] = w1;
    }
}

// ---------------------------------------------------------------------------
// Kernel 2: expert-grouped fused MLP.  grid.x = NEXP * 128 tile slots.
// 8 waves: wave = (mrow[0..1], nq[0..3]); up: each wave one 16x16 tile of the
// 32x64 ACT block; down: each wave 16 Y tiles (32x1024 Y in VGPRs).
// x tile (32x1024 bf16) is LDS-resident for the whole kernel; the up-weight
// panel is double-buffered so staging overlaps WMMA (1 barrier per K-step).
// ---------------------------------------------------------------------------
__global__ __launch_bounds__(256, 1) void moe_expert_kernel(
    const __bf16* __restrict__ xb,
    const float* __restrict__ wsh_up, const float* __restrict__ wsh_dn,
    const float* __restrict__ wrt_up, const float* __restrict__ wrt_dn,
    const int* __restrict__ cnt, const int* __restrict__ list,
    const float* __restrict__ wgt, float* __restrict__ out) {
    const int e    = blockIdx.x >> 7;
    const int tile = blockIdx.x & 127;
    const int nTok = cnt[e];
    if (tile * TM >= nTok) return;

    const float* up_r = wrt_up + (size_t)e * DIMD * HDIMD;
    const float* dn_r = wrt_dn + (size_t)e * DIMD * HDIMD;

    __shared__ __bf16 xbs[TM][1032];      // resident x tile, padded rows (66 KB)
    __shared__ __bf16 wpan[2][BNH][40];   // double-buffered up-weight panel
    __shared__ __bf16 acts[TM][72];       // relu^2 activations, bf16
    __shared__ int    s_tok[TM];
    __shared__ float  s_wt[TM];

    if (threadIdx.x < TM) {
        int s  = tile * TM + threadIdx.x;
        int ss = (s < nTok) ? s : (nTok - 1);
        s_tok[threadIdx.x] = list[e * NTOKS + ss];
        s_wt[threadIdx.x]  = (s < nTok) ? wgt[e * NTOKS + ss] : 0.0f;
    }
    __syncthreads();

    const int lane    = threadIdx.x & 31;
    const int wv      = threadIdx.x >> 5;
    const int mrow    = wv >> 2;               // 0..1
    const int nq      = wv & 3;                // 0..3
    const int lhalf8  = (lane >= 16) ? 8 : 0;  // C/D row half-select
    const int lhalf16 = (lane >= 16) ? 16 : 0; // A/B k half-select (k = 16h + s)
    const int lm      = lane & 15;

    // staging roles
    const int sw_k = threadIdx.x >> 6;         // 0..3
    const int sw_c = threadIdx.x & 63;         // 0..63

    // fill resident x tile: 256 threads x 128 bf16 each
    {
        int r  = threadIdx.x >> 3;
        int c0 = (threadIdx.x & 7) * 128;
        const bf16x8* src = (const bf16x8*)(xb + (size_t)s_tok[r] * DIMD + c0);
#pragma unroll
        for (int i = 0; i < 16; ++i)
            *(bf16x8*)(&xbs[r][c0 + i * 8]) = src[i];
    }
    // (covered by the first barrier inside the hb loop)

    f32x8 yacc[16];
#pragma unroll
    for (int j = 0; j < 16; ++j) yacc[j] = (f32x8)0.0f;

    for (int hb = 0; hb < HDIMD; hb += BNH) {
        // prologue: stage panel for kk = 0 into buffer 0
#pragma unroll
        for (int k2 = 0; k2 < 32; k2 += 4) {
            int k = sw_k + k2;
            size_t gi = (size_t)k * HDIMD + hb + sw_c;
            wpan[0][sw_c][k] = (__bf16)(wsh_up[gi] + up_r[gi]);
        }
        __syncthreads();

        f32x8 upacc = (f32x8)0.0f;
        for (int kk = 0; kk < DIMD; kk += 32) {
            const int buf = (kk >> 5) & 1;
            // stage next panel into the other buffer (overlaps with WMMA)
            if (kk + 32 < DIMD) {
#pragma unroll
                for (int k2 = 0; k2 < 32; k2 += 4) {
                    int k = sw_k + k2;
                    size_t gi = (size_t)(kk + 32 + k) * HDIMD + hb + sw_c;
                    wpan[buf ^ 1][sw_c][k] = (__bf16)(wsh_up[gi] + up_r[gi]);
                }
                if (kk + 64 < DIMD) {  // prefetch next-next -> global_prefetch_b8
                    size_t gn = (size_t)(kk + 64 + sw_k) * HDIMD + hb + sw_c;
                    __builtin_prefetch(wsh_up + gn, 0, 1);
                    __builtin_prefetch(up_r + gn, 0, 1);
                }
            }

            ABFrag A, B;
            A.h[0] = *(const bf16x8*)(&xbs[mrow * 16 + lm][kk + lhalf16]);
            A.h[1] = *(const bf16x8*)(&xbs[mrow * 16 + lm][kk + lhalf16 + 8]);
            B.h[0] = *(const bf16x8*)(&wpan[buf][nq * 16 + lm][lhalf16]);
            B.h[1] = *(const bf16x8*)(&wpan[buf][nq * 16 + lm][lhalf16 + 8]);
            upacc = __builtin_amdgcn_wmma_f32_16x16x32_bf16(
                false, A.v, false, B.v, (short)0, upacc, false, false);
            __syncthreads();
        }

        // relu^2 -> bf16 ACT tile (C/D layout: row = r + half8, col = lane&15)
        {
            int col   = nq * 16 + lm;
            int rbase = mrow * 16 + lhalf8;
#pragma unroll
            for (int r = 0; r < 8; ++r) {
                float v = upacc[r];
                v = (v > 0.0f) ? v * v : 0.0f;
                acts[rbase + r][col] = (__bf16)v;
            }
        }
        __syncthreads();

        // down: Y(32x1024) += ACT(32xBNH) @ (Wsh_dn + Wr_dn[e])[:, hb:hb+BNH]^T
#pragma unroll
        for (int j = 0; j < 16; ++j) {
            int dcol = (nq + 4 * j) * 16 + lm;
            const float* pws = wsh_dn + (size_t)dcol * HDIMD + hb;
            const float* pwr = dn_r + (size_t)dcol * HDIMD + hb;
#pragma unroll
            for (int ks = 0; ks < 2; ++ks) {
                int koff = ks * 32;
                ABFrag Aa, Bb;
                Aa.h[0] = *(const bf16x8*)(&acts[mrow * 16 + lm][koff + lhalf16]);
                Aa.h[1] = *(const bf16x8*)(&acts[mrow * 16 + lm][koff + lhalf16 + 8]);
                const f32x4* a4 = (const f32x4*)(pws + koff + lhalf16);
                const f32x4* b4 = (const f32x4*)(pwr + koff + lhalf16);
#pragma unroll
                for (int q = 0; q < 4; ++q) {
                    f32x4 s = a4[q] + b4[q];
#pragma unroll
                    for (int t = 0; t < 4; ++t) Bb.e[q * 4 + t] = (__bf16)s[t];
                }
                yacc[j] = __builtin_amdgcn_wmma_f32_16x16x32_bf16(
                    false, Aa.v, false, Bb.v, (short)0, yacc[j], false, false);
            }
        }
        __syncthreads();   // acts rewritten next iteration
    }

    // weighted scatter-add into out
#pragma unroll
    for (int j = 0; j < 16; ++j) {
        int dcol = (nq + 4 * j) * 16 + lm;
#pragma unroll
        for (int r = 0; r < 8; ++r) {
            int row  = mrow * 16 + r + lhalf8;
            float wt = s_wt[row];
            if (wt != 0.0f)
                atomicAdd(&out[(size_t)s_tok[row] * DIMD + dcol], wt * yacc[j][r]);
        }
    }
}

// ---------------------------------------------------------------------------
// Kernel 3: finalize scalar aux losses + tokens_per_expert
// ---------------------------------------------------------------------------
__global__ void moe_losses_kernel(const float* __restrict__ imp,
                                  const float* __restrict__ tpe,
                                  const float* __restrict__ zacc,
                                  float* __restrict__ tail) {
    if (threadIdx.x == 0 && blockIdx.x == 0) {
        const float Nf = (float)NTOKS;
        float lb = 0.0f;
        for (int e = 0; e < NEXP; ++e) lb += (tpe[e] / Nf) * (imp[e] / Nf);
        lb *= (float)NEXP;
        float z = zacc[0] / Nf;
        float mean = 0.0f;
        for (int e = 0; e < NEXP; ++e) mean += imp[e];
        mean /= (float)NEXP;
        float var = 0.0f;
        for (int e = 0; e < NEXP; ++e) { float d = imp[e] - mean; var += d * d; }
        var /= (float)(NEXP - 1);
        tail[0] = lb;
        tail[1] = z;
        tail[2] = var;
        for (int e = 0; e < NEXP; ++e) tail[3 + e] = tpe[e];
    }
}

// ---------------------------------------------------------------------------
extern "C" void kernel_launch(void* const* d_in, const int* in_sizes, int n_in,
                              void* d_out, int out_size, void* d_ws, size_t ws_size,
                              hipStream_t stream) {
    const float* x      = (const float*)d_in[0];
    const float* rw     = (const float*)d_in[1];
    const float* wsh_up = (const float*)d_in[2];
    const float* wsh_dn = (const float*)d_in[3];
    const float* wrt_up = (const float*)d_in[4];
    const float* wrt_dn = (const float*)d_in[5];
    float* out = (float*)d_out;

    char* ws = (char*)d_ws;
    __bf16* xb = (__bf16*)ws;                         // 4096*1024 bf16 = 8 MB
    size_t o   = (size_t)NTOKS * DIMD * sizeof(__bf16);
    int*   cnt  = (int*)(ws + o);                     // 8 ints
    float* tpe  = (float*)(ws + o + 64);              // 8 f32
    float* imp  = (float*)(ws + o + 128);             // 8 f32
    float* zacc = (float*)(ws + o + 192);             // 1 f32
    int*   list = (int*)(ws + o + 256);               // 8*4096 ints
    float* wgt  = (float*)(ws + o + 256 + (size_t)NEXP * NTOKS * sizeof(int));

    moe_zero_kernel<<<2048, 256, 0, stream>>>((f32x4*)out, (int*)(ws + o));
    moe_router_kernel<<<NTOKS / 8, 256, 0, stream>>>(x, rw, xb, cnt, list, wgt,
                                                     imp, tpe, zacc);
    moe_expert_kernel<<<NEXP * 128, 256, 0, stream>>>(xb, wsh_up, wsh_dn, wrt_up,
                                                      wrt_dn, cnt, list, wgt, out);
    moe_losses_kernel<<<1, 32, 0, stream>>>(imp, tpe, zacc,
                                            out + (size_t)NTOKS * DIMD);
}
